// Model_7404523618563
// MI455X (gfx1250) — matrix operand order repaired and checked by reading, not compiled, not run on hardware
//
#include <hip/hip_runtime.h>
#include <math.h>

// ---------------- model constants ----------------
#define BB     2
#define TT     512
#define CC     768
#define MM     (BB*TT)        // 1024 rows
#define HATT   12
#define NHEAD  64
#define HROSA  96
#define FFC    3072
#define NVOCAB 32000

typedef __attribute__((ext_vector_type(16))) __bf16 bf16x16;
typedef __attribute__((ext_vector_type(8)))  __bf16 bf16x8;
typedef __attribute__((ext_vector_type(8)))  float  floatx8;

__device__ __forceinline__ float sigmf(float x){ return 1.f/(1.f+__expf(-x)); }

// ================= GEMM: O[M,N] = A[M,K] @ W[K,N], f32 in/out, bf16 WMMA =================
// block = 256 thr = 8 waves; block tile 32(M) x 128(N); K-step 32; LDS-staged.
// Requires: M % 32 == 0, K % 32 == 0, N % 4 == 0 (true for all uses here).
// grid = (ceil(N/128), M/32)
#define LDA 40   // bf16 row stride for sA (16B-aligned fragment reads, padded)
#define LDB 40   // bf16 row stride for sB (N-major / transposed)
__global__ __launch_bounds__(256)
void k_gemm(const float* __restrict__ A, const float* __restrict__ W,
            float* __restrict__ O, int M, int N, int K) {
  __shared__ __bf16 sA[32*LDA];     // [mloc][k]   32x32 tile
  __shared__ __bf16 sB[128*LDB];    // [nloc][k]   32(K)x128(N) tile, N-major

  const int tid  = threadIdx.x;
  const int wave = tid >> 5;
  const int lane = tid & 31;
  const int tMb  = blockIdx.y * 32;
  const int tNb  = blockIdx.x * 128;

  const int mh = wave >> 2;              // 0/1 : which 16-row half
  const int nq = wave & 3;               // 0..3: which 32-col quarter
  const int l15 = lane & 15;
  const int kHiA = (lane >> 4) * 16;
  const int kHiB = (lane >> 4) * 16;
  const int mloc = mh*16 + l15;

  // fill assignments
  const int arow = tid >> 3;             // 0..31
  const int ac4  = (tid & 7) * 4;        // 0,4,...,28

  floatx8 acc0 = {}, acc1 = {};

  for (int kb = 0; kb < K; kb += 32) {
    __syncthreads();
    // ---- stage A tile: 32 rows x 32 k, one float4 per thread ----
    {
      const float* ap = A + (size_t)(tMb + arow)*K + kb + ac4;
      float4 av = *(const float4*)ap;
      __bf16* d = &sA[arow*LDA + ac4];
      d[0] = (__bf16)av.x; d[1] = (__bf16)av.y;
      d[2] = (__bf16)av.z; d[3] = (__bf16)av.w;
      if (kb + 32 < K) __builtin_prefetch(ap + 32, 0, 1);
    }
    // ---- stage B tile: 32 k x 128 n, transposed into sB[n][k]; 4 float4/thread ----
#pragma unroll
    for (int it = 0; it < 4; ++it) {
      int idx = tid + it*256;            // 0..1023
      int krow = idx >> 5;               // 0..31
      int c4   = (idx & 31) * 4;         // 0..124
      const float* bp = W + (size_t)(kb + krow)*N + tNb + c4;
      float bv[4];
      if (tNb + c4 + 3 < N) {
        float4 t = *(const float4*)bp;
        bv[0]=t.x; bv[1]=t.y; bv[2]=t.z; bv[3]=t.w;
        if (kb + 32 < K) __builtin_prefetch(bp + (size_t)32*N, 0, 1);
      } else {
#pragma unroll
        for (int j = 0; j < 4; ++j) {
          int n = tNb + c4 + j; if (n >= N) n = N-1;
          bv[j] = W[(size_t)(kb + krow)*N + n];
        }
      }
#pragma unroll
      for (int j = 0; j < 4; ++j)
        sB[(c4 + j)*LDB + krow] = (__bf16)bv[j];
    }
    __syncthreads();

    // ---- fragments from LDS (aligned b128 reads) ----
    bf16x16 af;
    {
      bf16x8 lo = *(const bf16x8*)&sA[mloc*LDA + kHiA];
      bf16x8 hi = *(const bf16x8*)&sA[mloc*LDA + kHiA + 8];
#pragma unroll
      for (int j = 0; j < 8; ++j) { af[j] = lo[j]; af[j+8] = hi[j]; }
    }
    bf16x16 bf0, bf1;
    {
      int n0 = nq*32 + l15;
      bf16x8 a0 = *(const bf16x8*)&sB[n0*LDB + kHiB];
      bf16x8 a1 = *(const bf16x8*)&sB[n0*LDB + kHiB + 8];
      int n1 = n0 + 16;
      bf16x8 b0 = *(const bf16x8*)&sB[n1*LDB + kHiB];
      bf16x8 b1 = *(const bf16x8*)&sB[n1*LDB + kHiB + 8];
#pragma unroll
      for (int j = 0; j < 8; ++j) {
        bf0[j] = a0[j]; bf0[j+8] = a1[j];
        bf1[j] = b0[j]; bf1[j+8] = b1[j];
      }
    }
    acc0 = __builtin_amdgcn_wmma_f32_16x16x32_bf16(false, af, false, bf0,
                                                   (short)0, acc0, false, false);
    acc1 = __builtin_amdgcn_wmma_f32_16x16x32_bf16(false, af, false, bf1,
                                                   (short)0, acc1, false, false);
  }

  // ---- store 2 x 16x16 tiles ----
  const int mrow = tMb + mh*16 + (lane >> 4)*8;
  {
    int n = tNb + nq*32 + l15;
    if (n < N) {
#pragma unroll
      for (int i = 0; i < 8; ++i) O[(size_t)(mrow+i)*N + n] = acc0[i];
    }
    n += 16;
    if (n < N) {
#pragma unroll
      for (int i = 0; i < 8; ++i) O[(size_t)(mrow+i)*N + n] = acc1[i];
    }
  }
}

// ================= embedding gather =================
__global__ void k_embed(const int* __restrict__ idx, const float* __restrict__ emb,
                        float* __restrict__ x) {
  int m = blockIdx.x;
  size_t src = (size_t)idx[m]*CC;
  for (int c = threadIdx.x; c < CC; c += blockDim.x)
    x[(size_t)m*CC + c] = emb[src + c];
}

// ================= LayerNorm over C=768, in-place safe =================
__global__ __launch_bounds__(256)
void k_layernorm(const float* __restrict__ x, const float* __restrict__ g,
                 const float* __restrict__ b, float* __restrict__ o) {
  int m = blockIdx.x;
  const float* row = x + (size_t)m*CC;
  float s = 0.f, s2 = 0.f;
  for (int c = threadIdx.x; c < CC; c += 256) { float v = row[c]; s += v; s2 += v*v; }
  for (int off = 16; off; off >>= 1) { s += __shfl_xor(s, off, 32); s2 += __shfl_xor(s2, off, 32); }
  __shared__ float ps[8], ps2[8];
  int lane = threadIdx.x & 31, wv = threadIdx.x >> 5;
  if (lane == 0) { ps[wv] = s; ps2[wv] = s2; }
  __syncthreads();
  if (threadIdx.x == 0) {
    float ts = 0.f, ts2 = 0.f;
    for (int i = 0; i < 8; ++i) { ts += ps[i]; ts2 += ps2[i]; }
    ps[0] = ts; ps2[0] = ts2;
  }
  __syncthreads();
  float mu  = ps[0] / (float)CC;
  float var = ps2[0] / (float)CC - mu*mu;
  float inv = rsqrtf(var + 1e-5f);
  for (int c = threadIdx.x; c < CC; c += 256) {
    float v = row[c];
    o[(size_t)m*CC + c] = (v - mu) * inv * g[c] + b[c];
  }
}

// ================= token-shift mix: o = x + (shift(x)-x)*coef =================
__global__ void k_mix(const float* __restrict__ x, const float* __restrict__ coef,
                      float* __restrict__ o) {
  int m = blockIdx.x;
  int t = m % TT;
  const float* row  = x + (size_t)m*CC;
  const float* prev = row - CC;       // valid only if t>0
  for (int c = threadIdx.x; c < CC; c += blockDim.x) {
    float xv = row[c];
    float pv = (t == 0) ? 0.f : prev[c];
    o[(size_t)m*CC + c] = xv + (pv - xv) * coef[c];
  }
}

// ================= generic elementwise =================
// mode 0: tanh(v*scale)  1: sigmoid(v*scale)  2: relu(v)^2
__global__ void k_act(float* __restrict__ p, int n, int mode, float scale) {
  int i = blockIdx.x*blockDim.x + threadIdx.x;
  if (i >= n) return;
  float v = p[i];
  if (mode == 0)      p[i] = tanhf(v*scale);
  else if (mode == 1) p[i] = sigmf(v*scale);
  else { float r = fmaxf(v, 0.f); p[i] = r*r; }
}

__global__ void k_add2(float* __restrict__ x, const float* __restrict__ a, int n) {
  int i = blockIdx.x*blockDim.x + threadIdx.x;
  if (i < n) x[i] += a[i];
}
__global__ void k_add3(float* __restrict__ x, const float* __restrict__ a,
                       const float* __restrict__ b, int n) {
  int i = blockIdx.x*blockDim.x + threadIdx.x;
  if (i < n) x[i] += a[i] + b[i];
}
__global__ void k_copy(float* __restrict__ d, const float* __restrict__ s, int n) {
  int i = blockIdx.x*blockDim.x + threadIdx.x;
  if (i < n) d[i] = s[i];
}

// w-decay: in-place: w = exp(-exp(-softplus(-(w0[c]+wa)) - 0.5))
__global__ void k_wdecay(float* __restrict__ wa, const float* __restrict__ w0, int n) {
  int i = blockIdx.x*blockDim.x + threadIdx.x;
  if (i >= n) return;
  int c = i % CC;
  float z  = -(w0[c] + wa[i]);
  float sp = fmaxf(z, 0.f) + log1pf(__expf(-fabsf(z)));
  float w  = -sp - 0.5f;
  wa[i] = __expf(-__expf(w));
}

// in-place: p = sigmoid(bias[c] + p)
__global__ void k_sigbias(float* __restrict__ p, const float* __restrict__ bias, int n) {
  int i = blockIdx.x*blockDim.x + threadIdx.x;
  if (i >= n) return;
  p[i] = sigmf(bias[i % CC] + p[i]);
}

// v = v + (vfirst - v) * sigmoid(v0[c] + vv)
__global__ void k_vmix(float* __restrict__ v, const float* __restrict__ vf,
                       const float* __restrict__ vv, const float* __restrict__ v0, int n) {
  int i = blockIdx.x*blockDim.x + threadIdx.x;
  if (i >= n) return;
  float s = sigmf(v0[i % CC] + vv[i]);
  v[i] += (vf[i] - v[i]) * s;
}

// ================= kk (L2-normalized per head) and k2 =================
// one wave per (m,h); 8 waves/block; grid = M*HATT/8
__global__ __launch_bounds__(256)
void k_kk_k2(const float* __restrict__ k, const float* __restrict__ a,
             const float* __restrict__ k_k, const float* __restrict__ k_a,
             float* __restrict__ kk, float* __restrict__ k2) {
  int idx  = blockIdx.x*8 + (threadIdx.x >> 5);
  int lane = threadIdx.x & 31;
  int m = idx / HATT, h = idx % HATT;
  size_t base = (size_t)m*CC + h*NHEAD;
  int c0 = h*NHEAD + lane, c1 = c0 + 32;
  float k0 = k[base+lane], k1 = k[base+32+lane];
  float kk0 = k0*k_k[c0], kk1 = k1*k_k[c1];
  float ss = kk0*kk0 + kk1*kk1;
  for (int off = 16; off; off >>= 1) ss += __shfl_xor(ss, off, 32);
  float inv = 1.f / fmaxf(sqrtf(ss), 1e-12f);
  kk[base+lane]    = kk0*inv;
  kk[base+32+lane] = kk1*inv;
  float a0 = a[base+lane], a1 = a[base+32+lane];
  k2[base+lane]    = k0*(1.f + (a0-1.f)*k_a[c0]);
  k2[base+32+lane] = k1*(1.f + (a1-1.f)*k_a[c1]);
}

// ================= RWKV-7 sequential scan =================
// one block (256 thr) per (b,h); state S[64][64] in regs: thr t owns row i=t>>2, cols (t&3)*16..+15
__global__ __launch_bounds__(256)
void k_rwkv_scan(const float* __restrict__ r, const float* __restrict__ wdec,
                 const float* __restrict__ k2, const float* __restrict__ v,
                 const float* __restrict__ kk, const float* __restrict__ ah,
                 float* __restrict__ y) {
  int bh = blockIdx.x;
  int b  = bh / HATT, h = bh % HATT;
  int i  = threadIdx.x >> 2;
  int jb = (threadIdx.x & 3) * 16;
  float S[16];
#pragma unroll
  for (int q = 0; q < 16; ++q) S[q] = 0.f;
  __shared__ float sr[64], sw[64], sk[64], sv[64], skk[64], sa[64];
  for (int t = 0; t < TT; ++t) {
    size_t base = ((size_t)(b*TT + t))*CC + h*NHEAD;
    __syncthreads();
    if (threadIdx.x < 64) {
      int c = threadIdx.x;
      sr[c] = r[base+c];  sw[c] = wdec[base+c]; sk[c] = k2[base+c];
      sv[c] = v[base+c];  skk[c] = kk[base+c];  sa[c] = ah[base+c];
    }
    __syncthreads();
    float sap = 0.f;
#pragma unroll
    for (int q = 0; q < 16; ++q) sap -= S[q] * skk[jb+q];   // a = -kk
    sap += __shfl_xor(sap, 1, 32);
    sap += __shfl_xor(sap, 2, 32);
    float vi = sv[i];
    float yp = 0.f;
#pragma unroll
    for (int q = 0; q < 16; ++q) {
      int j = jb + q;
      float bj = skk[j]*sa[j];                              // b = kk*a
      S[q] = S[q]*sw[j] + sap*bj + vi*sk[j];
      yp  += S[q]*sr[j];
    }
    yp += __shfl_xor(yp, 1, 32);
    yp += __shfl_xor(yp, 2, 32);
    if ((threadIdx.x & 3) == 0) y[base + i] = yp;
  }
}

// ================= fused GroupNorm + (r*k2*r_k).sum * v + gate =================
// one wave per (m,h); grid = M*HATT/8
__global__ __launch_bounds__(256)
void k_gn_rk_gate(const float* __restrict__ y, const float* __restrict__ gng,
                  const float* __restrict__ gnb, const float* __restrict__ r,
                  const float* __restrict__ k2, const float* __restrict__ rk,
                  const float* __restrict__ v, const float* __restrict__ g,
                  float* __restrict__ o) {
  int idx  = blockIdx.x*8 + (threadIdx.x >> 5);
  int lane = threadIdx.x & 31;
  int m = idx / HATT, h = idx % HATT;
  size_t base = (size_t)m*CC + h*NHEAD;
  float y0 = y[base+lane], y1 = y[base+32+lane];
  float s = y0 + y1, s2 = y0*y0 + y1*y1;
  float d0 = r[base+lane]*k2[base+lane]*rk[h*NHEAD+lane];
  float d1 = r[base+32+lane]*k2[base+32+lane]*rk[h*NHEAD+32+lane];
  float dd = d0 + d1;
  for (int off = 16; off; off >>= 1) {
    s  += __shfl_xor(s,  off, 32);
    s2 += __shfl_xor(s2, off, 32);
    dd += __shfl_xor(dd, off, 32);
  }
  float mu  = s / 64.f;
  float var = s2 / 64.f - mu*mu;
  float inv = rsqrtf(var + 64e-5f);
  int c0 = h*NHEAD + lane, c1 = c0 + 32;
  float yn0 = (y0-mu)*inv*gng[c0] + gnb[c0] + dd*v[base+lane];
  float yn1 = (y1-mu)*inv*gng[c1] + gnb[c1] + dd*v[base+32+lane];
  o[base+lane]    = yn0 * g[(size_t)m*CC + c0];
  o[base+32+lane] = yn1 * g[(size_t)m*CC + c1];
}

// ================= ROSA causal attention (soft bits), one wave per (b,h,t) =================
__global__ __launch_bounds__(256)
void k_rosa_attn(const float* __restrict__ qs, const float* __restrict__ ks,
                 const float* __restrict__ vb, const float* __restrict__ e0,
                 const float* __restrict__ e1, float* __restrict__ o) {
  int row  = blockIdx.x*8 + (threadIdx.x >> 5);
  int lane = threadIdx.x & 31;
  int t  = row % TT;
  int bh = row / TT;
  int h  = bh % HROSA, b = bh / HROSA;
  size_t qoff = ((size_t)(b*TT + t))*CC + h*8;
  float q[8];
#pragma unroll
  for (int d = 0; d < 8; ++d) q[d] = qs[qoff+d];
  const float scale = 0.3535533906f;            // 1/sqrt(8)
  float mx = -1e30f;
  for (int s = lane; s <= t; s += 32) {
    size_t ko = ((size_t)(b*TT + s))*CC + h*8;
    float sc = 0.f;
#pragma unroll
    for (int d = 0; d < 8; ++d) sc += q[d]*ks[ko+d];
    mx = fmaxf(mx, sc*scale);
  }
  for (int off = 16; off; off >>= 1) mx = fmaxf(mx, __shfl_xor(mx, off, 32));
  float den = 0.f, num[8];
#pragma unroll
  for (int d = 0; d < 8; ++d) num[d] = 0.f;
  for (int s = lane; s <= t; s += 32) {
    size_t ko = ((size_t)(b*TT + s))*CC + h*8;
    float sc = 0.f;
#pragma unroll
    for (int d = 0; d < 8; ++d) sc += q[d]*ks[ko+d];
    float e = __expf(sc*scale - mx);
    den += e;
#pragma unroll
    for (int d = 0; d < 8; ++d) num[d] += e * vb[ko+d];
  }
  for (int off = 16; off; off >>= 1) {
    den += __shfl_xor(den, off, 32);
#pragma unroll
    for (int d = 0; d < 8; ++d) num[d] += __shfl_xor(num[d], off, 32);
  }
  if (lane == 0) {
#pragma unroll
    for (int d = 0; d < 8; ++d) {
      float xo = num[d] / den;
      int c = h*8 + d;
      o[qoff+d] = e1[c]*xo + e0[c]*(1.f - xo);
    }
  }
}

// ======================================================================
// kernel_launch
// Input flattening assumption: jax tree_leaves order (dict keys ASCII-sorted):
//  0: idx
//  blocks[0]: att{Wk,Wo,Wr,Wv,a0,a1,a2,g1,g2,gn_b,gn_g,k_a,k_k,r_k,v0,v1,v2,
//                 w0,w1,w2,x_a,x_g,x_k,x_r,x_v,x_w} -> 1..26
//             ffn{Wk,Wv,x_k} -> 27..29 ; ln0_b,ln0_g,ln1_b,ln1_g,ln2_b,ln2_g,
//             ln3_b,ln3_g -> 30..37 ; rosa{Wk,Wq,Wv,emb0,emb1,x_k,x_q,x_v} -> 38..45
//  blocks[1]: att -> 46..71 ; ffn -> 72..74 ; ln1..ln3 -> 75..80 ; rosa -> 81..88
//  emb=89, head=90, ln_out_b=91, ln_out_g=92
// ======================================================================
extern "C" void kernel_launch(void* const* d_in, const int* in_sizes, int n_in,
                              void* d_out, int out_size, void* d_ws, size_t ws_size,
                              hipStream_t stream) {
  (void)in_sizes; (void)n_in; (void)out_size; (void)ws_size;
  const int* idx = (const int*)d_in[0];
  auto F = [&](int i) { return (const float*)d_in[i]; };

  // per-layer index bases
  const int attB[2]  = {1, 46};
  const int ffnB[2]  = {27, 72};
  const int lnB[2]   = {32, 75};   // ln1_b; +1 ln1_g; +2 ln2_b; ... +5 ln3_g
  const int rosaB[2] = {38, 81};
  // att offsets: Wk=0 Wo=1 Wr=2 Wv=3 a0=4 a1=5 a2=6 g1=7 g2=8 gn_b=9 gn_g=10
  //              k_a=11 k_k=12 r_k=13 v0=14 v1=15 v2=16 w0=17 w1=18 w2=19
  //              x_a=20 x_g=21 x_k=22 x_r=23 x_v=24 x_w=25
  // rosa offsets: Wk=0 Wq=1 Wv=2 emb0=3 emb1=4 x_k=5 x_q=6 x_v=7

  // workspace carve-up
  const size_t MC = (size_t)MM*CC;
  char* wsp = (char*)d_ws;
  float* buf[16];
  for (int i = 0; i < 16; ++i) buf[i] = (float*)(wsp + (size_t)i*MC*sizeof(float));
  float* fbuf   = (float*)(wsp + (size_t)16*MC*sizeof(float));     // M x 3072
  float* s64    = fbuf + (size_t)MM*FFC;                           // M x 64 (also used for x32)
  float* s128   = s64  + (size_t)MM*64;                            // M x 128
  float* vfirst = s128 + (size_t)MM*128;                           // M x C
  float* xres   = buf[15];

  const int EW = (int)((MC + 255) / 256);
  dim3 blk256(256);
  auto gemm = [&](const float* A, const float* W, float* O, int N, int K) {
    dim3 g((N + 127) / 128, MM / 32);
    k_gemm<<<g, blk256, 0, stream>>>(A, W, O, MM, N, K);
  };

  // ---------- embed + ln0 ----------
  k_embed<<<MM, blk256, 0, stream>>>(idx, F(89), xres);
  k_layernorm<<<MM, blk256, 0, stream>>>(xres, F(31), F(30), xres);   // ln0 (layer 0 only)

  for (int lid = 0; lid < 2; ++lid) {
    const int A0 = attB[lid], FF = ffnB[lid], LN = lnB[lid], RO = rosaB[lid];

    // ================= ROSA branch =================
    k_layernorm<<<MM, blk256, 0, stream>>>(xres, F(LN+5), F(LN+4), buf[0]);  // ln3
    k_mix<<<MM, blk256, 0, stream>>>(buf[0], F(RO+6), buf[1]);               // x_q
    k_mix<<<MM, blk256, 0, stream>>>(buf[0], F(RO+5), buf[2]);               // x_k
    k_mix<<<MM, blk256, 0, stream>>>(buf[0], F(RO+7), buf[3]);               // x_v
    gemm(buf[1], F(RO+1), buf[4], CC, CC);                                   // q
    gemm(buf[2], F(RO+0), buf[5], CC, CC);                                   // k
    gemm(buf[3], F(RO+2), buf[6], CC, CC);                                   // v
    k_act<<<EW, blk256, 0, stream>>>(buf[4], (int)MC, 0, 1000.f);            // tanh(q/ST)
    k_act<<<EW, blk256, 0, stream>>>(buf[5], (int)MC, 0, 1000.f);            // tanh(k/ST)
    k_act<<<EW, blk256, 0, stream>>>(buf[6], (int)MC, 1, 1000.f);            // sigmoid(v/ST)
    k_rosa_attn<<<BB*HROSA*TT/8, blk256, 0, stream>>>(buf[4], buf[5], buf[6],
                                                      F(RO+3), F(RO+4), buf[7]);

    // ================= time-mix branch =================
    k_layernorm<<<MM, blk256, 0, stream>>>(xres, F(LN+1), F(LN+0), buf[0]);  // ln1
    k_mix<<<MM, blk256, 0, stream>>>(buf[0], F(A0+23), buf[1]);              // x_r
    k_mix<<<MM, blk256, 0, stream>>>(buf[0], F(A0+25), buf[2]);              // x_w
    k_mix<<<MM, blk256, 0, stream>>>(buf[0], F(A0+22), buf[3]);              // x_k
    k_mix<<<MM, blk256, 0, stream>>>(buf[0], F(A0+24), buf[8]);              // x_v
    k_mix<<<MM, blk256, 0, stream>>>(buf[0], F(A0+20), buf[9]);              // x_a
    k_mix<<<MM, blk256, 0, stream>>>(buf[0], F(A0+21), buf[10]);             // x_g

    gemm(buf[1], F(A0+2), buf[11], CC, CC);                                  // r = xr@Wr
    // w lora: tanh(xw@w1)@w2 -> decay
    gemm(buf[2], F(A0+18), s64, 64, CC);
    k_act<<<(MM*64+255)/256, blk256, 0, stream>>>(s64, MM*64, 0, 1.f);
    gemm(s64, F(A0+19), buf[1], CC, 64);
    k_wdecay<<<EW, blk256, 0, stream>>>(buf[1], F(A0+17), (int)MC);          // buf1 = wdec

    gemm(buf[3], F(A0+0), buf[12], CC, CC);                                  // k = xk@Wk
    gemm(buf[8], F(A0+3), buf[13], CC, CC);                                  // v = xv@Wv
    if (lid == 0) {
      k_copy<<<EW, blk256, 0, stream>>>(vfirst, buf[13], (int)MC);
    } else {
      gemm(buf[8], F(A0+15), s64, 32, CC);                                   // xv@v1 (M x 32)
      gemm(s64, F(A0+16), buf[2], CC, 32);                                   // @v2
      k_vmix<<<EW, blk256, 0, stream>>>(buf[13], vfirst, buf[2], F(A0+14), (int)MC);
    }
    // a lora
    gemm(buf[9], F(A0+5), s64, 64, CC);
    gemm(s64, F(A0+6), buf[2], CC, 64);
    k_sigbias<<<EW, blk256, 0, stream>>>(buf[2], F(A0+4), (int)MC);          // buf2 = a
    // gate
    gemm(buf[10], F(A0+7), s128, 128, CC);
    k_act<<<(MM*128+255)/256, blk256, 0, stream>>>(s128, MM*128, 1, 1.f);
    gemm(s128, F(A0+8), buf[3], CC, 128);                                    // buf3 = g

    k_kk_k2<<<MM*HATT/8, blk256, 0, stream>>>(buf[12], buf[2], F(A0+12), F(A0+11),
                                              buf[8], buf[9]);               // kk=b8, k2=b9
    k_rwkv_scan<<<BB*HATT, blk256, 0, stream>>>(buf[11], buf[1], buf[9], buf[13],
                                                buf[8], buf[2], buf[10]);    // y=b10
    k_gn_rk_gate<<<MM*HATT/8, blk256, 0, stream>>>(buf[10], F(A0+10), F(A0+9),
                                                   buf[11], buf[9], F(A0+13),
                                                   buf[13], buf[3], buf[1]);
    gemm(buf[1], F(A0+1), buf[2], CC, CC);                                   // @Wo

    // residual: x += att_out + rosa_out
    k_add3<<<EW, blk256, 0, stream>>>(xres, buf[2], buf[7], (int)MC);

    // ================= channel-mix (FFN) =================
    k_layernorm<<<MM, blk256, 0, stream>>>(xres, F(LN+3), F(LN+2), buf[0]);  // ln2
    k_mix<<<MM, blk256, 0, stream>>>(buf[0], F(FF+2), buf[1]);               // x_k
    gemm(buf[1], F(FF+0), fbuf, FFC, CC);
    k_act<<<(MM*FFC+255)/256, blk256, 0, stream>>>(fbuf, MM*FFC, 2, 1.f);    // relu^2
    gemm(fbuf, F(FF+1), buf[2], CC, FFC);
    k_add2<<<EW, blk256, 0, stream>>>(xres, buf[2], (int)MC);
  }

  // ---------- final LN + head ----------
  k_layernorm<<<MM, blk256, 0, stream>>>(xres, F(92), F(91), buf[0]);
  {
    dim3 g((NVOCAB + 127) / 128, MM / 32);
    k_gemm<<<g, blk256, 0, stream>>>(buf[0], F(90), (float*)d_out, MM, NVOCAB, CC);
  }
}
